// HydraAttention_2310692405539
// MI455X (gfx1250) — matrix operand order repaired and checked by reading, not compiled, not run on hardware
//
#include <hip/hip_runtime.h>
#include <hip/hip_bf16.h>

// ---------------------------------------------------------------------------
// Hydra attention for MI455X (gfx1250, wave32, WMMA).
// All GEMM operands pre-split into hi/lo bf16 (fp32-class accuracy via
//   A*B ~= Ahi*Bhi + Ahi*Blo + Alo*Bhi on v_wmma_f32_16x16x32_bf16).
// GEMM main loop: async global->LDS copy (CDNA5 GLOBAL_LOAD_ASYNC_TO_LDS_B128
// via inline asm) + ds_load_b128 fragment reads + wmma.
// ---------------------------------------------------------------------------

typedef __bf16 v16bf __attribute__((ext_vector_type(16)));
typedef float  v8f   __attribute__((ext_vector_type(8)));

#define BT   16384   // B*T rows
#define DK   1024    // inner K dim
#define NQKV 3072    // qkv output cols
#define BM   128
#define BN   128
#define BKK  32

#define USE_ASYNC 1  // flip to 0 to fall back to sync int4 global->reg->LDS copy

// ---- bf16 bit helpers (RNE) ----
__device__ __forceinline__ unsigned short f2bf(float x) {
  unsigned int u = __float_as_uint(x);
  u += 0x7FFFu + ((u >> 16) & 1u);
  return (unsigned short)(u >> 16);
}
__device__ __forceinline__ float bf2f(unsigned short h) {
  return __uint_as_float(((unsigned int)h) << 16);
}
// split-convert 4 floats -> hi/lo bf16 quads (two 8B stores)
__device__ __forceinline__ void split4(float4 v, unsigned short* dh, unsigned short* dl) {
  union { unsigned short s[4]; uint2 u; } h, l;
  float f[4] = { v.x, v.y, v.z, v.w };
#pragma unroll
  for (int i = 0; i < 4; ++i) {
    unsigned short hi = f2bf(f[i]);
    h.s[i] = hi;
    l.s[i] = f2bf(f[i] - bf2f(hi));
  }
  *(uint2*)dh = h.u;
  *(uint2*)dl = l.u;
}

union FragU { int4 q[2]; v16bf v; };

// A fragment (16x32 bf16): lanes 0-15 -> M=lane, K {0..7,16..23};
// lanes 16-31 -> M=lane-16, K {8..15,24..31}.  Row stride = 32 shorts in LDS.
__device__ __forceinline__ v16bf fragA(const unsigned short* base, int lane) {
  const int m  = lane & 15;
  const int k0 = (lane < 16) ? 0 : 8;
  const unsigned short* p = base + m * BKK;
  FragU f;
  f.q[0] = *(const int4*)(p + k0);
  f.q[1] = *(const int4*)(p + k0 + 16);
  return f.v;
}
// B fragment (32x16 bf16), tile stored [n][k]: lanes 0-15 -> N=lane, K {0..7,16..23};
// lanes 16-31 -> N=lane-16, K {8..15,24..31}.
__device__ __forceinline__ v16bf fragB(const unsigned short* base, int lane) {
  const int n  = lane & 15;
  const int k0 = (lane < 16) ? 0 : 8;
  const unsigned short* p = base + n * BKK + k0;
  FragU f;
  f.q[0] = *(const int4*)(p);
  f.q[1] = *(const int4*)(p + 16);
  return f.v;
}

#if USE_ASYNC
// CDNA5 async tensor/data path: GLOBAL_LOAD_ASYNC_TO_LDS_B128 (GV mode).
// VDST = per-lane LDS byte address (low 32 bits of generic shared pointer),
// VADDR = 64-bit global address, SADDR = off.  Tracked by ASYNCcnt.
__device__ __forceinline__ void async_cp16(const unsigned short* g, unsigned short* l) {
  const unsigned int lds_off = (unsigned int)(unsigned long long)(const void*)l;
  asm volatile("global_load_async_to_lds_b128 %0, %1, off"
               :
               : "v"(lds_off), "v"((const void*)g)
               : "memory");
}
__device__ __forceinline__ void wait_async0() {
  asm volatile("s_wait_asynccnt 0x0" ::: "memory");
}
#endif

// ---------------------------------------------------------------------------
// C(BTxN) = A(BTx1024) * B(Nx1024)^T + bias ; optional fused row sumsq
// A,B given as pre-split hi/lo bf16 (raw bits in ushort), K-contiguous.
// ---------------------------------------------------------------------------
template <bool SUMSQ>
__global__ __launch_bounds__(256) void hydra_gemm_kernel(
    const unsigned short* __restrict__ Ahg, const unsigned short* __restrict__ Alg,
    const unsigned short* __restrict__ Bhg, const unsigned short* __restrict__ Blg,
    const float* __restrict__ bias, float* __restrict__ C, const int ldc,
    float* __restrict__ qss, float* __restrict__ kss) {
  __shared__ __align__(16) unsigned short LAh[2][BM * BKK];
  __shared__ __align__(16) unsigned short LAl[2][BM * BKK];
  __shared__ __align__(16) unsigned short LBh[2][BN * BKK];
  __shared__ __align__(16) unsigned short LBl[2][BN * BKK];

  const int tid  = threadIdx.x;
  const int lane = tid & 31;
  const int wave = tid >> 5;
  const int wm = (wave >> 1) * 32;   // 4 waves over M (32 rows each)
  const int wn = (wave & 1) * 64;    // 2 waves over N (64 cols each)
  const int n0 = blockIdx.x * BN;
  const int m0 = blockIdx.y * BM;

  // per-thread copy slice: 512 int4 per tile array, 2 per thread per array
  const int crow = tid >> 2;          // 0..63
  const int cc8  = (tid & 3) * 8;     // bf16 col offset within 32

  v8f acc[2][4] = {};

#if USE_ASYNC
  auto issueTile = [&](int buf, int bk) {
#pragma unroll
    for (int h = 0; h < 2; ++h) {
      const int row = crow + h * 64;
      const long ga = (long)(m0 + row) * DK + bk + cc8;
      const long gb = (long)(n0 + row) * DK + bk + cc8;
      const int lo = row * BKK + cc8;
      async_cp16(Ahg + ga, &LAh[buf][lo]);
      async_cp16(Alg + ga, &LAl[buf][lo]);
      async_cp16(Bhg + gb, &LBh[buf][lo]);
      async_cp16(Blg + gb, &LBl[buf][lo]);
    }
  };
#else
  int4 rAh[2], rAl[2], rBh[2], rBl[2];
  auto loadTile = [&](int bk) {
#pragma unroll
    for (int h = 0; h < 2; ++h) {
      const int row = crow + h * 64;
      const long ga = (long)(m0 + row) * DK + bk + cc8;
      const long gb = (long)(n0 + row) * DK + bk + cc8;
      rAh[h] = *(const int4*)(Ahg + ga);
      rAl[h] = *(const int4*)(Alg + ga);
      rBh[h] = *(const int4*)(Bhg + gb);
      rBl[h] = *(const int4*)(Blg + gb);
    }
  };
  auto storeTile = [&](int buf) {
#pragma unroll
    for (int h = 0; h < 2; ++h) {
      const int lo = (crow + h * 64) * BKK + cc8;
      *(int4*)&LAh[buf][lo] = rAh[h];
      *(int4*)&LAl[buf][lo] = rAl[h];
      *(int4*)&LBh[buf][lo] = rBh[h];
      *(int4*)&LBl[buf][lo] = rBl[h];
    }
  };
#endif

  const int KSTEPS = DK / BKK;  // 32

#if USE_ASYNC
  issueTile(0, 0);
#else
  loadTile(0);
  storeTile(0);
#endif

  for (int ks = 0; ks < KSTEPS; ++ks) {
    const int cur = ks & 1;
#if USE_ASYNC
    wait_async0();              // my async copies into buf 'cur' landed
    __syncthreads();            // everyone's landed
    if (ks + 1 < KSTEPS) issueTile(1 - cur, (ks + 1) * BKK);
#else
    __syncthreads();
    if (ks + 1 < KSTEPS) loadTile((ks + 1) * BKK);
#endif

    v16bf ah[2], al[2], bh[4], bl[4];
#pragma unroll
    for (int tm = 0; tm < 2; ++tm) {
      ah[tm] = fragA(&LAh[cur][(wm + tm * 16) * BKK], lane);
      al[tm] = fragA(&LAl[cur][(wm + tm * 16) * BKK], lane);
    }
#pragma unroll
    for (int tn = 0; tn < 4; ++tn) {
      bh[tn] = fragB(&LBh[cur][(wn + tn * 16) * BKK], lane);
      bl[tn] = fragB(&LBl[cur][(wn + tn * 16) * BKK], lane);
    }
#pragma unroll
    for (int tm = 0; tm < 2; ++tm)
#pragma unroll
      for (int tn = 0; tn < 4; ++tn) {
        acc[tm][tn] = __builtin_amdgcn_wmma_f32_16x16x32_bf16(
            false, ah[tm], false, bh[tn], (short)0, acc[tm][tn], false, false);
        acc[tm][tn] = __builtin_amdgcn_wmma_f32_16x16x32_bf16(
            false, ah[tm], false, bl[tn], (short)0, acc[tm][tn], false, false);
        acc[tm][tn] = __builtin_amdgcn_wmma_f32_16x16x32_bf16(
            false, al[tm], false, bh[tn], (short)0, acc[tm][tn], false, false);
      }

#if !USE_ASYNC
    if (ks + 1 < KSTEPS) storeTile(1 - cur);
#endif
  }

  // ---- epilogue: VGPR i -> M = i (lanes 0-15) / 8+i (lanes 16-31), N = lane&15
  float biasv[4];
#pragma unroll
  for (int tn = 0; tn < 4; ++tn)
    biasv[tn] = bias[n0 + wn + tn * 16 + (lane & 15)];

  const int region = SUMSQ ? (n0 >> 10) : 2;  // 0=q,1=k,2=v
  float* ssp = (region == 0) ? qss : kss;

#pragma unroll
  for (int tm = 0; tm < 2; ++tm) {
#pragma unroll
    for (int i = 0; i < 8; ++i) {
      const int rloc = wm + tm * 16 + ((lane < 16) ? i : (8 + i));
      const long row = (long)(m0 + rloc);
      const int col0 = n0 + wn + (lane & 15);
      float cv[4];
      float s = 0.f;
#pragma unroll
      for (int tn = 0; tn < 4; ++tn) {
        cv[tn] = acc[tm][tn][i] + biasv[tn];
        C[row * ldc + col0 + tn * 16] = cv[tn];
        s += cv[tn] * cv[tn];
      }
      if (SUMSQ && region < 2) {
        s += __shfl_xor(s, 1);
        s += __shfl_xor(s, 2);
        s += __shfl_xor(s, 4);
        s += __shfl_xor(s, 8);
        if ((lane & 15) == 0) atomicAdd(&ssp[row], s);
      }
    }
  }
}

// split f32 array into hi/lo bf16 arrays (vectorized x4)
__global__ __launch_bounds__(256) void hydra_split_kernel(
    const float* __restrict__ src, unsigned short* __restrict__ dh,
    unsigned short* __restrict__ dl, int n4) {
  const int f = blockIdx.x * blockDim.x + threadIdx.x;
  if (f < n4) {
    const float4 v = *(const float4*)(src + f * 4);
    split4(v, dh + f * 4, dl + f * 4);
  }
}

// GEMM2 A-operand: a[r,d] = q[r,d] * qscale[r] * kv[b,d]  -> hi/lo bf16
__global__ __launch_bounds__(256) void hydra_scale_split_kernel(
    const float* __restrict__ qkv, const float* __restrict__ qsc,
    const float* __restrict__ kv, unsigned short* __restrict__ dh,
    unsigned short* __restrict__ dl) {
  const int f = blockIdx.x * blockDim.x + threadIdx.x;  // [0, BT*DK/4)
  const int row = f >> 8;          // 256 float4 per row
  const int d4 = (f & 255) * 4;
  float4 v = *(const float4*)(qkv + (long)row * NQKV + d4);
  const float qs = qsc[row];
  const float4 k4 = *(const float4*)(kv + ((row >> 12) << 10) + d4);
  v = make_float4(v.x * qs * k4.x, v.y * qs * k4.y, v.z * qs * k4.z, v.w * qs * k4.w);
  const long o = (long)row * DK + d4;
  split4(v, dh + o, dl + o);
}

// kv[b,d] = sum_t (mask ? 0 : k*rsqrt(sumsq_k)*v)
__global__ __launch_bounds__(256) void hydra_kv_kernel(
    const float* __restrict__ qkv, const int* __restrict__ mask,
    const float* __restrict__ kss, float* __restrict__ kv) {
  const int b = blockIdx.x >> 2;
  const int d = ((blockIdx.x & 3) << 8) + threadIdx.x;
  const int t0 = blockIdx.y * 128;
  float acc = 0.f;
  for (int tt = 0; tt < 128; ++tt) {
    const int row = b * 4096 + t0 + tt;
    if (mask[row] == 0) {
      const float sc = __frsqrt_rn(kss[row]);
      const long base = (long)row * NQKV;
      acc += qkv[base + 1024 + d] * sc * qkv[base + 2048 + d];
    }
  }
  atomicAdd(&kv[b * 1024 + d], acc);
}

__global__ void hydra_qscale_kernel(const float* __restrict__ qss,
                                    float* __restrict__ qscale) {
  const int i = blockIdx.x * blockDim.x + threadIdx.x;
  if (i < BT) qscale[i] = __frsqrt_rn(qss[i]);
}

__global__ void hydra_zero_kernel(float* __restrict__ p, int n) {
  const int i = blockIdx.x * blockDim.x + threadIdx.x;
  if (i < n) p[i] = 0.f;
}

extern "C" void kernel_launch(void* const* d_in, const int* in_sizes, int n_in,
                              void* d_out, int out_size, void* d_ws, size_t ws_size,
                              hipStream_t stream) {
  const float* x    = (const float*)d_in[0];
  const int*   mask = (const int*)d_in[1];
  const float* Wqkv = (const float*)d_in[2];
  const float* bqkv = (const float*)d_in[3];
  const float* Wout = (const float*)d_in[4];
  const float* bout = (const float*)d_in[5];
  float* out = (float*)d_out;

  // ---- workspace layout ----
  float* ws  = (float*)d_ws;
  float* qkv = ws;                          // 16384*3072 f32
  float* qss = ws + (long)BT * NQKV;        // 16384
  float* kss = qss + BT;                    // 16384
  float* kv  = kss + BT;                    // 4096
  float* qsc = kv + 4 * DK;                 // 16384
  unsigned short* us = (unsigned short*)(qsc + BT);  // 16B-aligned
  unsigned short* xh  = us;                 // 16M bf16
  unsigned short* xl  = xh + (long)BT * DK;
  unsigned short* wqh = xl + (long)BT * DK; // 3M
  unsigned short* wql = wqh + (long)NQKV * DK;
  unsigned short* woh = wql + (long)NQKV * DK;  // 1M
  unsigned short* wol = woh + (long)DK * DK;
  // GEMM2 A operand aliases x's split arrays (x dead after GEMM1)
  unsigned short* a2h = xh;
  unsigned short* a2l = xl;

  hydra_zero_kernel<<<(2 * BT + 4 * DK + 255) / 256, 256, 0, stream>>>(
      qss, 2 * BT + 4 * DK);

  // pre-split operands to hi/lo bf16
  hydra_split_kernel<<<(BT * DK / 4 + 255) / 256, 256, 0, stream>>>(x, xh, xl, BT * DK / 4);
  hydra_split_kernel<<<(NQKV * DK / 4 + 255) / 256, 256, 0, stream>>>(Wqkv, wqh, wql, NQKV * DK / 4);
  hydra_split_kernel<<<(DK * DK / 4 + 255) / 256, 256, 0, stream>>>(Wout, woh, wol, DK * DK / 4);

  // GEMM1: qkv projection + fused sumsq(q), sumsq(k)
  hydra_gemm_kernel<true><<<dim3(NQKV / BN, BT / BM), 256, 0, stream>>>(
      xh, xl, wqh, wql, bqkv, qkv, NQKV, qss, kss);

  hydra_qscale_kernel<<<BT / 256, 256, 0, stream>>>(qss, qsc);
  hydra_kv_kernel<<<dim3(16, 32), 256, 0, stream>>>(qkv, mask, kss, kv);

  // build scaled GEMM2 A operand: q * qscale * kv  -> hi/lo bf16
  hydra_scale_split_kernel<<<BT * DK / 4 / 256, 256, 0, stream>>>(qkv, qsc, kv, a2h, a2l);

  // GEMM2: out = A2 @ Wout^T + bout
  hydra_gemm_kernel<false><<<dim3(DK / BN, BT / BM), 256, 0, stream>>>(
      a2h, a2l, woh, wol, bout, out, DK, nullptr, nullptr);
}
